// PNL_48808008352101
// MI455X (gfx1250) — matrix operand order repaired and checked by reading, not compiled, not run on hardware
//
#include <hip/hip_runtime.h>

// ---------------------------------------------------------------------------
// Problem constants (fixed by setup_inputs)
// ---------------------------------------------------------------------------
#define BB 8
#define NN 2048
#define CCH 768
#define DDL 384
#define MALL (BB * NN)          // 16384 tokens
#define BN_EPS 1e-5f

typedef __attribute__((ext_vector_type(16))) __bf16 v16bf;
typedef __attribute__((ext_vector_type(8)))  float  v8f;

// ---------------------------------------------------------------------------
// fp32 -> bf16 convert (vectorized by 4)
// ---------------------------------------------------------------------------
__global__ __launch_bounds__(256) void f32_to_bf16_kernel(
    const float* __restrict__ in, __bf16* __restrict__ out, int n4) {
  int i = blockIdx.x * 256 + threadIdx.x;
  if (i >= n4) return;
  float4 v = reinterpret_cast<const float4*>(in)[i];
  union { __bf16 h[4]; uint2 u; } o;
  o.h[0] = (__bf16)v.x; o.h[1] = (__bf16)v.y;
  o.h[2] = (__bf16)v.z; o.h[3] = (__bf16)v.w;
  reinterpret_cast<uint2*>(out)[i] = o.u;
}

// ---------------------------------------------------------------------------
// Batched 32x32 LDS tile transpose: in [R x Cc] -> out [Cc x R] (bf16)
// block = (32, 8)
// ---------------------------------------------------------------------------
__global__ __launch_bounds__(256) void transpose_bf16_kernel(
    const __bf16* __restrict__ in, __bf16* __restrict__ out,
    int R, int Cc, long long inStride, long long outStride) {
  __shared__ __bf16 tile[32][33];
  const __bf16* src = in  + (size_t)blockIdx.z * inStride;
  __bf16*       dst = out + (size_t)blockIdx.z * outStride;
  int c0 = blockIdx.x * 32, r0 = blockIdx.y * 32;
  int tx = threadIdx.x, ty = threadIdx.y;
#pragma unroll
  for (int i = 0; i < 4; i++)
    tile[ty + i * 8][tx] = src[(size_t)(r0 + ty + i * 8) * Cc + c0 + tx];
  __syncthreads();
#pragma unroll
  for (int i = 0; i < 4; i++)
    dst[(size_t)(c0 + ty + i * 8) * R + r0 + tx] = tile[tx][ty + i * 8];
}

// ---------------------------------------------------------------------------
// WMMA fragment loaders (bf16, wave32 layouts per CDNA5 ISA 7.12.2)
// A 16x32 (MxK): lane = half*16 + m ; elements 0..7 = K[half*8 .. half*8+7],
//                elements 8..15 = K[16+half*8 .. 16+half*8+7]
// B 32x16 (KxN): lane = half*16 + n ; elements 0..7 = K[half*8 .. half*8+7],
//                elements 8..15 = K[16+half*8 .. 16+half*8+7]
// ---------------------------------------------------------------------------
__device__ __forceinline__ v16bf load_a_frag(const __bf16* p) {
  union { uint4 u[2]; v16bf v; } r;
  r.u[0] = *reinterpret_cast<const uint4*>(p);
  r.u[1] = *reinterpret_cast<const uint4*>(p + 16);
  return r.v;
}
__device__ __forceinline__ v16bf load_b_frag(const __bf16* p) {
  union { uint4 u[2]; v16bf v; } r;
  r.u[0] = *reinterpret_cast<const uint4*>(p);
  r.u[1] = *reinterpret_cast<const uint4*>(p + 16);
  return r.v;
}

// ---------------------------------------------------------------------------
// Generic batched NT GEMM:  out[m,n] = alpha * sum_k A[m,k]*B[n,k] + bias[n]
// A: [M x K] bf16 row-major (lda), B: [N x K] bf16 row-major (ldb)
// out: bf16 or fp32 (ldc).
//
// block = 256 threads = 8 waves arranged 4(M) x 2(N); wave tile 32x64
// (2 A-frags x 4 B-frags = 8 accumulators); block tile 128x128.
// Fragment sets are ping-pong double-buffered across 32-deep k-phases so the
// 12 b128 loads of phase p+1 overlap the 8 WMMAs of phase p.
// Requires M%128==0, N%128==0, K%64==0 (all shapes here comply).
// ---------------------------------------------------------------------------
template <bool OUT_BF16>
__global__ __launch_bounds__(256) void gemm_nt_wmma_kernel(
    const __bf16* __restrict__ A, const __bf16* __restrict__ B, void* Cptr,
    const float* __restrict__ bias, int K, int lda, int ldb, int ldc,
    long long strideA, long long strideB, long long strideC, float alpha) {
  const int wave = threadIdx.x >> 5;
  const int lane = threadIdx.x & 31;
  const int half = lane >> 4;
  const int lm   = lane & 15;
  const int wm   = wave & 3;   // wave position along M (4)
  const int wn   = wave >> 2;  // wave position along N (2)

  const int m0 = blockIdx.y * 128 + wm * 32;
  const int n0 = blockIdx.x * 128 + wn * 64;

  A += (size_t)blockIdx.z * strideA;
  B += (size_t)blockIdx.z * strideB;

  v8f acc[2][4] = {{v8f{}, v8f{}, v8f{}, v8f{}}, {v8f{}, v8f{}, v8f{}, v8f{}}};

  const __bf16* Ar0 = A + (size_t)(m0 + lm) * lda + half * 8;
  const __bf16* Ar1 = Ar0 + (size_t)16 * lda;
  const __bf16* Bp0 = B + (size_t)(n0 + 0 * 16 + lm) * ldb + half * 8;
  const __bf16* Bp1 = B + (size_t)(n0 + 1 * 16 + lm) * ldb + half * 8;
  const __bf16* Bp2 = B + (size_t)(n0 + 2 * 16 + lm) * ldb + half * 8;
  const __bf16* Bp3 = B + (size_t)(n0 + 3 * 16 + lm) * ldb + half * 8;

  auto load_set = [&](v16bf* af, v16bf* bf_, int k) {
    af[0]  = load_a_frag(Ar0 + k);
    af[1]  = load_a_frag(Ar1 + k);
    bf_[0] = load_b_frag(Bp0 + k);
    bf_[1] = load_b_frag(Bp1 + k);
    bf_[2] = load_b_frag(Bp2 + k);
    bf_[3] = load_b_frag(Bp3 + k);
  };
  auto do_wmma = [&](const v16bf* af, const v16bf* bf_) {
#pragma unroll
    for (int r = 0; r < 2; r++)
#pragma unroll
      for (int j = 0; j < 4; j++)
        acc[r][j] = __builtin_amdgcn_wmma_f32_16x16x32_bf16(
            false, af[r], false, bf_[j], (short)0, acc[r][j], false, false);
  };

  v16bf aP[2], bP[4], aQ[2], bQ[4];
  load_set(aP, bP, 0);
  for (int k0 = 0; k0 < K; k0 += 64) {
    load_set(aQ, bQ, k0 + 32);          // phase B loads in flight
    do_wmma(aP, bP);                    // compute phase A
    if (k0 + 64 < K) load_set(aP, bP, k0 + 64);  // phase A' loads in flight
    do_wmma(aQ, bQ);                    // compute phase B
  }

  // C/D layout: VGPR i -> row m0 + r*16 + half*8 + i, col n0 + j*16 + lm
#pragma unroll
  for (int r = 0; r < 2; r++) {
#pragma unroll
    for (int j = 0; j < 4; j++) {
      const int col = n0 + j * 16 + lm;
      const float bv = bias ? bias[col] : 0.0f;
#pragma unroll
      for (int i = 0; i < 8; i++) {
        const int row = m0 + r * 16 + half * 8 + i;
        const float v = alpha * acc[r][j][i] + bv;
        if (OUT_BF16) {
          ((__bf16*)Cptr)[(size_t)blockIdx.z * strideC +
                          (size_t)row * ldc + col] = (__bf16)v;
        } else {
          ((float*)Cptr)[(size_t)blockIdx.z * strideC +
                         (size_t)row * ldc + col] = v;
        }
      }
    }
  }
}

// ---------------------------------------------------------------------------
// BatchNorm: zero stats, partial sums (256 rows per block, 3 channels/thread),
// then finalize + residual.
// ---------------------------------------------------------------------------
__global__ __launch_bounds__(256) void zero_kernel(float* p, int n) {
  int i = blockIdx.x * 256 + threadIdx.x;
  if (i < n) p[i] = 0.0f;
}

__global__ __launch_bounds__(256) void bn_stats_kernel(
    const float* __restrict__ y, float* __restrict__ stats) {
  const int c  = threadIdx.x;                 // 0..255
  const int r0 = blockIdx.x * 256;            // 64 blocks * 256 rows = 16384
  float s0 = 0.f, s1 = 0.f, s2 = 0.f;
  float q0 = 0.f, q1 = 0.f, q2 = 0.f;
  for (int r = 0; r < 256; r++) {
    const float* row = y + (size_t)(r0 + r) * CCH;
    float v0 = row[c], v1 = row[c + 256], v2 = row[c + 512];
    s0 += v0; q0 += v0 * v0;
    s1 += v1; q1 += v1 * v1;
    s2 += v2; q2 += v2 * v2;
  }
  atomicAdd(&stats[c],              s0);
  atomicAdd(&stats[c + 256],        s1);
  atomicAdd(&stats[c + 512],        s2);
  atomicAdd(&stats[CCH + c],        q0);
  atomicAdd(&stats[CCH + c + 256],  q1);
  atomicAdd(&stats[CCH + c + 512],  q2);
}

__global__ __launch_bounds__(256) void bn_finalize_kernel(
    float* __restrict__ y, const float* __restrict__ xh,
    const float* __restrict__ stats, const float* __restrict__ gamma,
    const float* __restrict__ beta, long long n) {
  long long i = (long long)blockIdx.x * 256 + threadIdx.x;
  if (i >= n) return;
  const int c = (int)(i % CCH);
  const float invCnt = 1.0f / (float)MALL;
  const float mean = stats[c] * invCnt;
  const float var  = stats[CCH + c] * invCnt - mean * mean;
  const float inv  = rsqrtf(var + BN_EPS);
  y[i] = gamma[c] * (y[i] - mean) * inv + beta[c] + xh[i];
}

// ---------------------------------------------------------------------------
// Host-side launch
// ---------------------------------------------------------------------------
extern "C" void kernel_launch(void* const* d_in, const int* in_sizes, int n_in,
                              void* d_out, int out_size, void* d_ws, size_t ws_size,
                              hipStream_t stream) {
  const float* x_h     = (const float*)d_in[0];
  const float* x_l     = (const float*)d_in[1];
  const float* w_g     = (const float*)d_in[2];
  const float* b_g     = (const float*)d_in[3];
  const float* w_theta = (const float*)d_in[4];
  const float* b_theta = (const float*)d_in[5];
  const float* w_phi   = (const float*)d_in[6];
  const float* b_phi   = (const float*)d_in[7];
  const float* w_out   = (const float*)d_in[8];
  const float* b_out   = (const float*)d_in[9];
  const float* gamma   = (const float*)d_in[10];
  const float* beta    = (const float*)d_in[11];
  float* out = (float*)d_out;

  char* ws = (char*)d_ws;
  size_t off = 0;
  auto carve = [&](size_t bytes) {
    void* p = ws + off;
    off += (bytes + 255) & ~(size_t)255;
    return p;
  };

  __bf16* xh_bf   = (__bf16*)carve((size_t)MALL * CCH * 2);
  __bf16* xl_bf   = (__bf16*)carve((size_t)MALL * CCH * 2);
  __bf16* wth_bf  = (__bf16*)carve((size_t)DDL * CCH * 2);
  __bf16* wph_bf  = (__bf16*)carve((size_t)DDL * CCH * 2);
  __bf16* wg_bf   = (__bf16*)carve((size_t)DDL * CCH * 2);
  __bf16* wout_bf = (__bf16*)carve((size_t)CCH * DDL * 2);
  __bf16* Theta   = (__bf16*)carve((size_t)MALL * DDL * 2);   // [B*N, D]
  __bf16* Phi     = (__bf16*)carve((size_t)MALL * DDL * 2);
  __bf16* Gm      = (__bf16*)carve((size_t)MALL * DDL * 2);
  __bf16* PhiT    = (__bf16*)carve((size_t)MALL * DDL * 2);   // [B][D][N]
  __bf16* GT      = (__bf16*)carve((size_t)MALL * DDL * 2);
  __bf16* Smat    = (__bf16*)carve((size_t)BB * DDL * DDL * 2); // [B][D][D]
  __bf16* Mt      = (__bf16*)carve((size_t)BB * CCH * DDL * 2); // [B][C][D]
  float*  stats   = (float*)carve((size_t)2 * CCH * 4);

  // ---- fp32 -> bf16 conversions ----
  {
    int n4 = MALL * CCH / 4;
    f32_to_bf16_kernel<<<(n4 + 255) / 256, 256, 0, stream>>>(x_h, xh_bf, n4);
    f32_to_bf16_kernel<<<(n4 + 255) / 256, 256, 0, stream>>>(x_l, xl_bf, n4);
    int w4 = DDL * CCH / 4;
    f32_to_bf16_kernel<<<(w4 + 255) / 256, 256, 0, stream>>>(w_theta, wth_bf, w4);
    f32_to_bf16_kernel<<<(w4 + 255) / 256, 256, 0, stream>>>(w_phi, wph_bf, w4);
    f32_to_bf16_kernel<<<(w4 + 255) / 256, 256, 0, stream>>>(w_g, wg_bf, w4);
    f32_to_bf16_kernel<<<(w4 + 255) / 256, 256, 0, stream>>>(w_out, wout_bf, w4);
  }

  // ---- projections: [16384 x 768] x [384 x 768]^T -> bf16 [16384 x 384] ----
  {
    dim3 grid(DDL / 128, MALL / 128, 1);
    gemm_nt_wmma_kernel<true><<<grid, 256, 0, stream>>>(
        xh_bf, wth_bf, Theta, b_theta, CCH, CCH, CCH, DDL, 0, 0, 0, 1.0f);
    gemm_nt_wmma_kernel<true><<<grid, 256, 0, stream>>>(
        xl_bf, wph_bf, Phi, b_phi, CCH, CCH, CCH, DDL, 0, 0, 0, 1.0f);
    gemm_nt_wmma_kernel<true><<<grid, 256, 0, stream>>>(
        xl_bf, wg_bf, Gm, b_g, CCH, CCH, CCH, DDL, 0, 0, 0, 1.0f);
  }

  // ---- per-batch transpose [N x D] -> [D x N] for Phi, G ----
  {
    dim3 tb(32, 8);
    dim3 grid(DDL / 32, NN / 32, BB);
    long long st = (long long)NN * DDL;
    transpose_bf16_kernel<<<grid, tb, 0, stream>>>(Phi, PhiT, NN, DDL, st, st);
    transpose_bf16_kernel<<<grid, tb, 0, stream>>>(Gm, GT, NN, DDL, st, st);
  }

  // ---- S_b = (Phi_b^T G_b) / N : A=PhiT [D x N], B=GT [D x N] -> bf16 [D x D]
  {
    dim3 grid(DDL / 128, DDL / 128, BB);
    gemm_nt_wmma_kernel<true><<<grid, 256, 0, stream>>>(
        PhiT, GT, Smat, nullptr, NN, NN, NN, DDL,
        (long long)DDL * NN, (long long)DDL * NN, (long long)DDL * DDL,
        1.0f / (float)NN);
  }

  // ---- Mt_b[c,d] = sum_e w_out[c,e] * S_b[d,e]  (folds output projection) --
  {
    dim3 grid(DDL / 128, CCH / 128, BB);
    gemm_nt_wmma_kernel<true><<<grid, 256, 0, stream>>>(
        wout_bf, Smat, Mt, nullptr, DDL, DDL, DDL, DDL,
        0, (long long)DDL * DDL, (long long)CCH * DDL, 1.0f);
  }

  // ---- Y_b = Theta_b * Mt_b^T + b_out -> fp32 into d_out [B][N][C] ----
  {
    dim3 grid(CCH / 128, NN / 128, BB);
    gemm_nt_wmma_kernel<false><<<grid, 256, 0, stream>>>(
        Theta, Mt, out, b_out, DDL, DDL, DDL, CCH,
        (long long)NN * DDL, (long long)CCH * DDL, (long long)NN * CCH, 1.0f);
  }

  // ---- BatchNorm (training-mode stats over B*N per channel) + residual ----
  zero_kernel<<<(2 * CCH + 255) / 256, 256, 0, stream>>>(stats, 2 * CCH);
  bn_stats_kernel<<<MALL / 256, 256, 0, stream>>>(out, stats);
  {
    long long n = (long long)MALL * CCH;
    bn_finalize_kernel<<<(unsigned)((n + 255) / 256), 256, 0, stream>>>(
        out, x_h, stats, gamma, beta, n);
  }
}
